// SelfAttention_30073361007353
// MI455X (gfx1250) — matrix operand order repaired and checked by reading, not compiled, not run on hardware
//
#include <hip/hip_runtime.h>
#include <stdint.h>

// ---------------- problem constants ----------------
#define MODEL_DIM 1024
#define NUM_HEADS 16
#define HEAD_DIM  64
#define SEQ_T     2048
#define BATCH     4
#define ROWS      (BATCH * SEQ_T)     // 8192
#define QKV_N     (3 * MODEL_DIM)     // 3072

// ---------------- CDNA5 WMMA types ----------------
typedef __attribute__((ext_vector_type(16))) __bf16       v16bf;
typedef __attribute__((ext_vector_type(8)))  float        v8f;
typedef __attribute__((ext_vector_type(4)))  unsigned int v4u;

union FragB16 {
  v16bf          v;
  unsigned int   u[8];
  v4u            q[2];
  unsigned short s[16];
};

__device__ __forceinline__ v8f wmma_bf16(v16bf a, v16bf b, v8f c) {
  // D = A(16x32 bf16) * B(32x16 bf16) + C(16x16 f32)
  return __builtin_amdgcn_wmma_f32_16x16x32_bf16(false, a, false, b,
                                                 (short)0, c, false, false);
}

__device__ __forceinline__ unsigned short f2bf(float f) {
  unsigned u = __float_as_uint(f);
  u += 0x7FFFu + ((u >> 16) & 1u);            // round-to-nearest-even
  return (unsigned short)(u >> 16);
}

// Async DMA: memory -> LDS, 16 bytes per lane, tracked by ASYNCcnt.
// ldsOff: wave-relative LDS byte address (low 32 bits of generic shared ptr).
// gOff:   byte offset against uniform 64-bit SGPR base.
__device__ __forceinline__ void async_g2l_b128(unsigned ldsOff, unsigned gOff,
                                               const void* sbase) {
  asm volatile("global_load_async_to_lds_b128 %0, %1, %2"
               :: "v"(ldsOff), "v"(gOff), "s"(sbase) : "memory");
}
__device__ __forceinline__ void wait_asynccnt0() {
  asm volatile("s_wait_asynccnt 0" ::: "memory");
}

// ---------------- fp32 -> bf16 convert ----------------
__global__ void cvt_bf16_kernel(const float* __restrict__ src,
                                unsigned short* __restrict__ dst, int n) {
  int i = blockIdx.x * blockDim.x + threadIdx.x;
  int stride = gridDim.x * blockDim.x;
  for (; i < n; i += stride) dst[i] = f2bf(src[i]);
}

// ---------------- tiled bf16 WMMA GEMM ----------------
// C[M,N] = A[M,K] * B[K,N] + bias
// Block tile 128x128, K-step 32, 256 threads = 8 waves arranged 4(m) x 2(n).
// Each wave: 32x64 output = 2x4 tiles of 16x16 (8 v8f accumulators).
// MODE 0: scatter bf16 q/k/v head-major.  MODE 1: fp32 output row-major.
template <int MODE>
__global__ __launch_bounds__(256)
void gemm_bf16_wmma(const unsigned short* __restrict__ A,   // [M x K] bf16
                    const unsigned short* __restrict__ Bw,  // [K x N] bf16
                    const float* __restrict__ bias,         // [N] fp32
                    unsigned short* __restrict__ qh,
                    unsigned short* __restrict__ kh,
                    unsigned short* __restrict__ vh,
                    float* __restrict__ outf,
                    int M, int N, int K) {
  __shared__ unsigned short As[128 * 40];   // [row][k], pad 8 -> 80B rows (16B aligned)
  __shared__ unsigned short Bs[128 * 40];   // [n][k] transposed

  const int tid   = threadIdx.x;
  const int lane  = tid & 31;
  const int wave  = tid >> 5;
  const int wm    = wave >> 1;     // 0..3
  const int wn    = wave & 1;      // 0..1
  const int m0    = blockIdx.y * 128;
  const int n0    = blockIdx.x * 128;
  const int row16 = lane & 15;
  const int hi    = lane >> 4;     // lane half
  const int kb8   = hi * 8;        // half the A-frag K base per lane half (base = 2*kb8)

  v8f acc[2][4];
  for (int mt = 0; mt < 2; ++mt)
    for (int nt = 0; nt < 4; ++nt)
      for (int e = 0; e < 8; ++e) acc[mt][nt][e] = 0.0f;

  const int ksteps = K >> 5;
  for (int kb = 0; kb < ksteps; ++kb) {
    __syncthreads();
    // Stage A tile via async DMA: 128 rows x 64B = 512 b128 transfers, 2/thread.
    for (int i = 0; i < 2; ++i) {
      int id = tid + i * 256;               // 0..511
      int r  = id >> 2, q4 = id & 3;
      unsigned gOff = (unsigned)((((size_t)(m0 + r)) * K + kb * 32 + q4 * 8) * 2);
      unsigned lOff = (unsigned)(uintptr_t)&As[r * 40 + q4 * 8];
      async_g2l_b128(lOff, gOff, A);
    }
    // Stage B tile transposed: Bs[n][k] = Bw[kb*32+k][n0+n] (coalesced in n)
    for (int idx = tid; idx < 32 * 128; idx += 256) {
      int k = idx >> 7, n = idx & 127;
      Bs[n * 40 + k] = Bw[(size_t)(kb * 32 + k) * N + n0 + n];
    }
    // Prefetch next B tile rows (emits global_prefetch_b8 on gfx1250)
    {
      int kn = (kb + 1 < ksteps) ? kb + 1 : kb;
      __builtin_prefetch(Bw + (size_t)(kn * 32 + (tid >> 3)) * N + n0 + (tid & 7) * 16, 0, 1);
    }
    wait_asynccnt0();
    __syncthreads();

    // A fragments: 16x32, lane half picks K offset {0,16}; VGPR0-7: contiguous K 2*kb8+0..15
    FragB16 a[2];
    for (int mt = 0; mt < 2; ++mt) {
      const unsigned short* base = &As[(wm * 32 + mt * 16 + row16) * 40];
      for (int j = 0; j < 4; ++j) {
        a[mt].u[j]     = *(const unsigned*)(base + 2 * kb8 + 2 * j);
        a[mt].u[j + 4] = *(const unsigned*)(base + 8 + 2 * kb8 + 2 * j);
      }
    }
    // B fragments: 32x16, lane = column, contiguous K (hi*16..hi*16+15) -> two b128 LDS loads
    FragB16 b[4];
    for (int nt = 0; nt < 4; ++nt) {
      const unsigned short* base = &Bs[(wn * 64 + nt * 16 + row16) * 40 + hi * 16];
      b[nt].q[0] = *(const v4u*)(base);
      b[nt].q[1] = *(const v4u*)(base + 8);
    }
    for (int mt = 0; mt < 2; ++mt)
      for (int nt = 0; nt < 4; ++nt)
        acc[mt][nt] = wmma_bf16(a[mt].v, b[nt].v, acc[mt][nt]);
  }

  // Epilogue. C layout: lanes 0-15: M=r, lanes 16-31: M=r+8; N = lane&15.
  for (int mt = 0; mt < 2; ++mt) {
    for (int nt = 0; nt < 4; ++nt) {
      int col  = n0 + wn * 64 + nt * 16 + row16;
      float bv = bias[col];
      for (int r = 0; r < 8; ++r) {
        int   rowg = m0 + wm * 32 + mt * 16 + hi * 8 + r;
        float v    = acc[mt][nt][r] + bv;
        if (MODE == 0) {
          // scatter to head-major bf16 q/k/v: [b*H+h][t][hd]
          int which = col >> 10;
          int rem   = col & 1023;
          int h = rem >> 6, hd = rem & 63;
          int bb = rowg >> 11, t = rowg & 2047;
          size_t dst = ((size_t)(bb * NUM_HEADS + h) * SEQ_T + t) * HEAD_DIM + hd;
          unsigned short bf = f2bf(v);
          unsigned short* p = (which == 0) ? qh : (which == 1) ? kh : vh;
          p[dst] = bf;
        } else {
          outf[(size_t)rowg * N + col] = v;
        }
      }
    }
  }
}

// ---------------- flash attention (causal, bf16 WMMA, f32 softmax) ----------------
// grid: (T/64, B*H). 128 threads = 4 waves; wave w owns 16 q rows.
__global__ __launch_bounds__(128)
void attn_kernel(const unsigned short* __restrict__ qh,
                 const unsigned short* __restrict__ kh,
                 const unsigned short* __restrict__ vh,
                 unsigned short* __restrict__ ao) {
  __shared__ unsigned short Ks[64 * 72];      // [key][hd]   (B-frags for Q*K^T)
  __shared__ unsigned short Vs[64 * 72];      // [hd][key]   (B-frags for P*V)
  __shared__ unsigned short Ps[4 * 16 * 72];  // per-wave P tile, C->A relayout

  const int tid   = threadIdx.x;
  const int lane  = tid & 31;
  const int w     = tid >> 5;
  const int row16 = lane & 15;
  const int hi    = lane >> 4;
  const int kb8   = hi * 8;
  const int bx    = blockIdx.x;               // q tile of 64 rows
  const int bh    = blockIdx.y;               // b*H + h
  const int qrow0 = bx * 64 + w * 16;

  // Q fragments (A-layout 16x32, two K-steps covering hd 0..63), loaded once.
  FragB16 qf[2];
  {
    const unsigned short* qp =
        qh + ((size_t)bh * SEQ_T + qrow0 + row16) * HEAD_DIM;
    for (int f = 0; f < 2; ++f)
      for (int j = 0; j < 4; ++j) {
        qf[f].u[j]     = *(const unsigned*)(qp + f * 32 + 2 * kb8 + 2 * j);
        qf[f].u[j + 4] = *(const unsigned*)(qp + f * 32 + 8 + 2 * kb8 + 2 * j);
      }
  }

  float m[8], l[8];
  v8f o[4];
  for (int r = 0; r < 8; ++r) { m[r] = -3.0e38f; l[r] = 0.0f; }
  for (int n = 0; n < 4; ++n)
    for (int e = 0; e < 8; ++e) o[n][e] = 0.0f;

  unsigned short* Pw = &Ps[w * 16 * 72];

  for (int kt = 0; kt <= bx; ++kt) {          // causal: only lower tiles
    const int k0 = kt * 64;
    __syncthreads();
    // Stage K via async DMA: 64 keys x 128B = 512 b128 transfers, 4/thread.
    for (int i = 0; i < 4; ++i) {
      int id  = tid + i * 128;                // 0..511
      int key = id >> 3, q8 = id & 7;
      unsigned gOff =
          (unsigned)((((size_t)bh * SEQ_T + k0 + key) * HEAD_DIM + q8 * 8) * 2);
      unsigned lOff = (unsigned)(uintptr_t)&Ks[key * 72 + q8 * 8];
      async_g2l_b128(lOff, gOff, kh);
    }
    // Stage V transposed: [hd][key] (dword global loads, 2 LDS b16 stores)
    for (int idx = tid; idx < 64 * 32; idx += 128) {
      int key = idx >> 5, hd2 = idx & 31;
      unsigned v = *(const unsigned*)(
          vh + ((size_t)bh * SEQ_T + k0 + key) * HEAD_DIM + hd2 * 2);
      Vs[(hd2 * 2 + 0) * 72 + key] = (unsigned short)(v & 0xFFFFu);
      Vs[(hd2 * 2 + 1) * 72 + key] = (unsigned short)(v >> 16);
    }
    wait_asynccnt0();
    __syncthreads();

    // scores S = Q * K^T : 4 column tiles of 16 keys, 2 K-steps (hd) each
    v8f s[4];
    for (int n = 0; n < 4; ++n) {
      const unsigned short* kbase = &Ks[(n * 16 + row16) * 72 + hi * 16];
      FragB16 b0, b1;
      b0.q[0] = *(const v4u*)(kbase);
      b0.q[1] = *(const v4u*)(kbase + 8);
      b1.q[0] = *(const v4u*)(kbase + 32);
      b1.q[1] = *(const v4u*)(kbase + 40);
      v8f z;
      for (int e = 0; e < 8; ++e) z[e] = 0.0f;
      z = wmma_bf16(qf[0].v, b0.v, z);
      z = wmma_bf16(qf[1].v, b1.v, z);
      s[n] = z;
    }

    // scale + causal mask (diagonal tile only)
    const bool diag = (kt == bx);
    for (int n = 0; n < 4; ++n)
      for (int r = 0; r < 8; ++r) {
        float v = s[n][r] * 0.125f;            // 1/sqrt(64)
        if (diag) {
          int key = k0 + n * 16 + row16;
          int qr  = qrow0 + hi * 8 + r;
          if (key > qr) v = -3.0e38f;
        }
        s[n][r] = v;
      }

    // online softmax: row stats live per 16-lane half (lanes share rows)
    float al[8];
    for (int r = 0; r < 8; ++r) {
      float c = fmaxf(fmaxf(s[0][r], s[1][r]), fmaxf(s[2][r], s[3][r]));
      c = fmaxf(c, __shfl_xor(c, 1));
      c = fmaxf(c, __shfl_xor(c, 2));
      c = fmaxf(c, __shfl_xor(c, 4));
      c = fmaxf(c, __shfl_xor(c, 8));
      float mn = fmaxf(m[r], c);
      al[r] = __expf(m[r] - mn);
      m[r]  = mn;
    }
    float rs[8];
    for (int r = 0; r < 8; ++r) rs[r] = 0.0f;
    for (int n = 0; n < 4; ++n)
      for (int r = 0; r < 8; ++r) {
        float p = __expf(s[n][r] - m[r]);
        rs[r] += p;
        Pw[(hi * 8 + r) * 72 + n * 16 + row16] = f2bf(p);   // C->A relayout via LDS
      }
    for (int r = 0; r < 8; ++r) {
      float t = rs[r];
      t += __shfl_xor(t, 1);
      t += __shfl_xor(t, 2);
      t += __shfl_xor(t, 4);
      t += __shfl_xor(t, 8);
      l[r] = l[r] * al[r] + t;
    }
    for (int n = 0; n < 4; ++n)
      for (int r = 0; r < 8; ++r) o[n][r] *= al[r];

    __syncthreads();   // P visible in A-layout reads (uniform across block)

    // O += P * V : P as A-frags (K = key), V as B-frags from [hd][key]
    FragB16 pf[2];
    {
      const unsigned short* pp = Pw + row16 * 72;
      for (int f = 0; f < 2; ++f)
        for (int j = 0; j < 4; ++j) {
          pf[f].u[j]     = *(const unsigned*)(pp + f * 32 + 2 * kb8 + 2 * j);
          pf[f].u[j + 4] = *(const unsigned*)(pp + f * 32 + 8 + 2 * kb8 + 2 * j);
        }
    }
    for (int n = 0; n < 4; ++n) {
      const unsigned short* vb = &Vs[(n * 16 + row16) * 72 + hi * 16];
      FragB16 b0, b1;
      b0.q[0] = *(const v4u*)(vb);
      b0.q[1] = *(const v4u*)(vb + 8);
      b1.q[0] = *(const v4u*)(vb + 32);
      b1.q[1] = *(const v4u*)(vb + 40);
      o[n] = wmma_bf16(pf[0].v, b0.v, o[n]);
      o[n] = wmma_bf16(pf[1].v, b1.v, o[n]);
    }
  }

  // normalize + store bf16 to [B,T,D] layout for out-projection
  const int b = bh >> 4, h = bh & 15;
  for (int n = 0; n < 4; ++n) {
    int hd = n * 16 + row16;
    for (int r = 0; r < 8; ++r) {
      int   qr = qrow0 + hi * 8 + r;
      float v  = o[n][r] / l[r];
      ao[((size_t)(b * SEQ_T + qr)) * MODEL_DIM + h * HEAD_DIM + hd] = f2bf(v);
    }
  }
}

// ---------------- launcher ----------------
extern "C" void kernel_launch(void* const* d_in, const int* in_sizes, int n_in,
                              void* d_out, int out_size, void* d_ws, size_t ws_size,
                              hipStream_t stream) {
  const float* x    = (const float*)d_in[0];
  const float* Wqkv = (const float*)d_in[1];
  const float* bqkv = (const float*)d_in[2];
  const float* Wout = (const float*)d_in[3];
  const float* bout = (const float*)d_in[4];
  float* out = (float*)d_out;

  char*  ws  = (char*)d_ws;
  size_t off = 0;
  auto alloc = [&](size_t bytes) {
    char* p = ws + off;
    off += (bytes + 255) & ~(size_t)255;
    return p;
  };
  unsigned short* xbf  = (unsigned short*)alloc((size_t)ROWS * MODEL_DIM * 2);
  unsigned short* wqbf = (unsigned short*)alloc((size_t)MODEL_DIM * QKV_N * 2);
  unsigned short* wobf = (unsigned short*)alloc((size_t)MODEL_DIM * MODEL_DIM * 2);
  unsigned short* qhb  = (unsigned short*)alloc((size_t)ROWS * MODEL_DIM * 2);
  unsigned short* khb  = (unsigned short*)alloc((size_t)ROWS * MODEL_DIM * 2);
  unsigned short* vhb  = (unsigned short*)alloc((size_t)ROWS * MODEL_DIM * 2);
  unsigned short* aob  = (unsigned short*)alloc((size_t)ROWS * MODEL_DIM * 2);
  (void)ws_size; (void)in_sizes; (void)n_in; (void)out_size;

  cvt_bf16_kernel<<<512, 256, 0, stream>>>(x, xbf, ROWS * MODEL_DIM);
  cvt_bf16_kernel<<<512, 256, 0, stream>>>(Wqkv, wqbf, MODEL_DIM * QKV_N);
  cvt_bf16_kernel<<<256, 256, 0, stream>>>(Wout, wobf, MODEL_DIM * MODEL_DIM);

  gemm_bf16_wmma<0><<<dim3(QKV_N / 128, ROWS / 128), 256, 0, stream>>>(
      xbf, wqbf, bqkv, qhb, khb, vhb, nullptr, ROWS, QKV_N, MODEL_DIM);

  attn_kernel<<<dim3(SEQ_T / 64, BATCH * NUM_HEADS), 128, 0, stream>>>(
      qhb, khb, vhb, aob);

  gemm_bf16_wmma<1><<<dim3(MODEL_DIM / 128, ROWS / 128), 256, 0, stream>>>(
      aob, wobf, bout, nullptr, nullptr, nullptr, out, ROWS, MODEL_DIM, MODEL_DIM);
}
